// EntropyGuidedAttention_84293028151600
// MI455X (gfx1250) — matrix operand order repaired and checked by reading, not compiled, not run on hardware
//
#include <hip/hip_runtime.h>
#include <hip/hip_bf16.h>
#include <math.h>
#include <stdint.h>

typedef __attribute__((ext_vector_type(16))) __bf16 v16bf;
typedef __attribute__((ext_vector_type(2)))  __bf16 v2bf;
typedef __attribute__((ext_vector_type(2)))  float  v2f;
typedef __attribute__((ext_vector_type(8)))  float  v8f;

#define WMMA_BF16(a, b, c) \
  __builtin_amdgcn_wmma_f32_16x16x32_bf16(false, (a), false, (b), (short)0, (c), false, false)

constexpr int Bsz  = 2;
constexpr int Nseq = 2048;
constexpr int Cdim = 1024;
constexpr int Hn   = 16;
constexpr int Dh   = 64;
constexpr int QKVN = 3 * Cdim;                                  // 3072
constexpr size_t QKV_S_STRIDE = (size_t)Bsz * Hn * Nseq * Dh;   // elems per s in qkv ws

// ------------- vectorized fragment loads (2 x ds_load_b128 each) ------------

__device__ inline v16bf frag_pair(const __bf16* p0, const __bf16* p1) {
  union { uint4 q[2]; v16bf v; } u;
  u.q[0] = *(const uint4*)p0;
  u.q[1] = *(const uint4*)p1;
  return u.v;
}

// A-matrix 16x32 fragment from row-major tile row (= &T[m][kbase]).
__device__ inline v16bf frag_a_row(const __bf16* row, int half) {
  return frag_pair(row + 8 * half, row + 16 + 8 * half);
}

// B-matrix 32x16 fragment from TRANSPOSED tile row (= &Bt[n][kbase]).
__device__ inline v16bf frag_b_row(const __bf16* nrow, int half) {
  return frag_pair(nrow + 8 * half, nrow + 16 + 8 * half);
}

// packed pair conversion -> single v_cvt_pk_bf16_f32
__device__ inline uint32_t pack2bf(float a, float b) {
  v2f f = {a, b};
  union { v2bf h; uint32_t u; } cv;
  cv.h = __builtin_convertvector(f, v2bf);
  return cv.u;
}

// ---------------- kernel 1: entropy gate  ew = sigmoid(x @ We + be) ---------

__global__ __launch_bounds__(256)
void gate_kernel(const float* __restrict__ x, const float* __restrict__ We,
                 const float* __restrict__ be, float* __restrict__ ew) {
  int idx = blockIdx.x * 256 + threadIdx.x;      // total B*N*H = 65536
  int bn = idx >> 4;
  int h  = idx & 15;
  const float* xr = x + (size_t)bn * Cdim;
  float s = be[h];
  for (int c = 0; c < Cdim; ++c) s += xr[c] * We[c * Hn + h];
  ew[(size_t)bn * Hn + h] = 1.0f / (1.0f + __expf(-s));
}

// ---------------- kernel 2: QKV projection (bf16 WMMA GEMM) -----------------
// grid = (3072/64, 4096/64), block = 128 (4 waves). Software-pipelined k-loop.

__global__ __launch_bounds__(128)
void qkv_gemm(const float* __restrict__ x, const float* __restrict__ Wqkv,
              const float* __restrict__ bqkv, __bf16* __restrict__ qkv) {
  __shared__ __align__(16) __bf16 As [64][32];   // row-major A tile
  __shared__ __align__(16) __bf16 Bst[64][32];   // TRANSPOSED B tile: Bst[n][k]

  int tid = threadIdx.x;
  int lane = tid & 31, w = tid >> 5;
  int lane16 = lane & 15, half = lane >> 4;
  int cblk = blockIdx.x * 64;
  int rblk = blockIdx.y * 64;

  int a_rb = tid >> 4, a_c2 = (tid & 15) * 2;
  int b_c  = tid & 63, b_r2 = tid >> 6;

  const float* apg = &x[(size_t)(rblk + a_rb) * Cdim + a_c2];
  const float* bpg = &Wqkv[(size_t)(2 * b_r2) * QKVN + cblk + b_c];

  v8f acc[4] = {};

  // prologue: load tile k0 = 0 into registers
  float2 ta[8];
  float  tb0[8], tb1[8];
#pragma unroll
  for (int it = 0; it < 8; ++it)
    ta[it] = *(const float2*)&apg[(size_t)it * 8 * Cdim];
#pragma unroll
  for (int it = 0; it < 8; ++it) {
    tb0[it] = bpg[(size_t)(4 * it) * QKVN];
    tb1[it] = bpg[(size_t)(4 * it + 1) * QKVN];
  }

  for (int k0 = 0; k0 < Cdim; k0 += 32) {
    __syncthreads();
#pragma unroll
    for (int it = 0; it < 8; ++it)
      *(uint32_t*)&As[a_rb + it * 8][a_c2] = pack2bf(ta[it].x, ta[it].y);
#pragma unroll
    for (int it = 0; it < 8; ++it)
      *(uint32_t*)&Bst[b_c][2 * b_r2 + 4 * it] = pack2bf(tb0[it], tb1[it]);
    __syncthreads();

    // issue next tile's global loads; latency hides under the WMMAs below
    if (k0 + 32 < Cdim) {
      int kn = k0 + 32;
#pragma unroll
      for (int it = 0; it < 8; ++it)
        ta[it] = *(const float2*)&apg[kn + (size_t)it * 8 * Cdim];
#pragma unroll
      for (int it = 0; it < 8; ++it) {
        tb0[it] = bpg[(size_t)(kn + 4 * it) * QKVN];
        tb1[it] = bpg[(size_t)(kn + 4 * it + 1) * QKVN];
      }
    }

    v16bf af = frag_a_row(&As[w * 16 + lane16][0], half);
    v16bf bf[4];
#pragma unroll
    for (int nt = 0; nt < 4; ++nt)
      bf[nt] = frag_b_row(&Bst[nt * 16 + lane16][0], half);
#pragma unroll
    for (int nt = 0; nt < 4; ++nt)
      acc[nt] = WMMA_BF16(af, bf[nt], acc[nt]);
  }

  // epilogue: block-uniform decomposition, constant-offset stores
  int sS = cblk >> 10;                 // which of q/k/v
  int hh = (cblk & 1023) >> 6;         // head
  int bb = rblk >> 11;                 // batch
  int nb = (rblk & 2047) + w * 16 + 8 * half;
  __bf16* qp = qkv + (size_t)sS * QKV_S_STRIDE + (size_t)bb * (Hn * Nseq * Dh) +
               (size_t)hh * (Nseq * Dh) + (size_t)nb * Dh;
#pragma unroll
  for (int nt = 0; nt < 4; ++nt) {
    int d = nt * 16 + lane16;
    float bias = bqkv[cblk + d];
#pragma unroll
    for (int i = 0; i < 8; ++i)
      qp[(size_t)i * Dh + d] = (__bf16)(acc[nt][i] + bias);
  }
}

// ---------------- kernel 3: flash attention with entropy gate ---------------
// grid = B*H * (N/64) blocks, 128 threads. Software-pipelined key-chunk loop.

__global__ __launch_bounds__(128)
void attn_kernel(const __bf16* __restrict__ qkv, const float* __restrict__ ew,
                 const int* __restrict__ am, __bf16* __restrict__ aw) {
  __shared__ __align__(16) __bf16 Qs [64][64];   // row-major [q][d]   (A operand)
  __shared__ __align__(16) __bf16 Ks [64][64];   // row-major [key][d] (B^T for S)
  __shared__ __align__(16) __bf16 Vst[64][64];   // TRANSPOSED [d][key] (B for P@V)
  __shared__ __align__(16) __bf16 Ps [4][16][64];// per-wave P staging  (A operand)

  int tid = threadIdx.x;
  int lane = tid & 31, w = tid >> 5;
  int lane16 = lane & 15, half = lane >> 4;

  int bid = blockIdx.x;
  int qt = bid & 31;                             // N/64 = 32 query tiles
  int bh = bid >> 5;
  int h = bh & 15, b = bh >> 4;
  int qbase = qt * 64;

  const __bf16* Qg = qkv + (size_t)b * (Hn * Nseq * Dh) + (size_t)h * (Nseq * Dh);
  const __bf16* Kg = Qg + QKV_S_STRIDE;
  const __bf16* Vg = Kg + QKV_S_STRIDE;

  { // load 64x64 bf16 Q tile (512 x uint4, 4 per thread)
    const uint4* src = (const uint4*)(Qg + (size_t)qbase * Dh);
    uint4* dst = (uint4*)&Qs[0][0];
#pragma unroll
    for (int it = 0; it < 4; ++it) dst[it * 128 + tid] = src[it * 128 + tid];
  }
  __syncthreads();

  v16bf qa[2];
  qa[0] = frag_a_row(&Qs[w * 16 + lane16][0], half);
  qa[1] = frag_a_row(&Qs[w * 16 + lane16][32], half);

  float mstate[8], lstate[8], g[8];
  const float* gp = &ew[(size_t)(b * Nseq + qbase + w * 16 + 8 * half) * Hn + h];
#pragma unroll
  for (int i = 0; i < 8; ++i) {
    mstate[i] = -__builtin_inff();
    lstate[i] = 0.0f;
    g[i] = gp[i * Hn];
  }
  v8f O[4] = {};

  const float scl = 0.125f;                      // 1/sqrt(Dh)
  const int* mbase = am + (size_t)b * Nseq * Nseq +
                     (size_t)(qbase + w * 16 + 8 * half) * Nseq + lane16;

  // fixed per-thread staging coordinates
  const uint4* kptr = (const uint4*)Kg + tid;                 // K: direct copy
  int v_kb = tid >> 3, v_d0 = (tid & 7) * 8;                  // V: transpose
  const uint4* vptr = (const uint4*)(Vg + (size_t)v_kb * Dh + v_d0);

  // prologue: preload chunk 0 into registers
  uint4 tk[4], tv[4];
#pragma unroll
  for (int it = 0; it < 4; ++it) {
    tk[it] = kptr[it * 128];
    tv[it] = vptr[it * 128];
  }

  for (int j = 0; j < Nseq / 64; ++j) {
    int kb = j * 64;
    __syncthreads();
    { // commit staged registers to LDS
      uint4* kdst = (uint4*)&Ks[0][0];
#pragma unroll
      for (int it = 0; it < 4; ++it) kdst[it * 128 + tid] = tk[it];
#pragma unroll
      for (int it = 0; it < 4; ++it) {
        union { uint4 q; __bf16 hh[8]; } u;
        u.q = tv[it];
#pragma unroll
        for (int e = 0; e < 8; ++e) Vst[v_d0 + e][v_kb + it * 16] = u.hh[e];
      }
    }
    __syncthreads();

    // issue next chunk's global loads; latency hides under WMMA work below
    if (j + 1 < Nseq / 64) {
      int base = (j + 1) * 512;                  // 64 keys * 64 d / 8 per uint4
#pragma unroll
      for (int it = 0; it < 4; ++it) {
        tk[it] = kptr[base + it * 128];
        tv[it] = vptr[base + it * 128];
      }
    }

    // S = Q @ K^T : per 16-key tile, batch both B fragments then 2 WMMAs
    v8f S[4];
#pragma unroll
    for (int nt = 0; nt < 4; ++nt) {
      v16bf bf0 = frag_b_row(&Ks[nt * 16 + lane16][0], half);
      v16bf bf1 = frag_b_row(&Ks[nt * 16 + lane16][32], half);
      v8f z = {};
      z = WMMA_BF16(qa[0], bf0, z);
      z = WMMA_BF16(qa[1], bf1, z);
      S[nt] = z;
    }

    // mask values for this chunk (issued early; latency overlaps WMMAs)
    const int* mchunk = mbase + kb;
    int mv[8][4];
#pragma unroll
    for (int i = 0; i < 8; ++i)
#pragma unroll
      for (int nt = 0; nt < 4; ++nt)
        mv[i][nt] = mchunk[(size_t)i * Nseq + nt * 16];

    // gate/scale/mask + online softmax (row = i + 8*half across 16 lanes)
#pragma unroll
    for (int i = 0; i < 8; ++i) {
      float mx = -__builtin_inff();
#pragma unroll
      for (int nt = 0; nt < 4; ++nt) {
        float lv = S[nt][i] * scl * g[i];
        if (mv[i][nt] == 0) lv = -__builtin_inff();
        S[nt][i] = lv;
        mx = fmaxf(mx, lv);
      }
      mx = fmaxf(mx, __shfl_xor(mx, 1, 32));
      mx = fmaxf(mx, __shfl_xor(mx, 2, 32));
      mx = fmaxf(mx, __shfl_xor(mx, 4, 32));
      mx = fmaxf(mx, __shfl_xor(mx, 8, 32));
      float mnew  = fmaxf(mstate[i], mx);
      bool  dead  = (mnew == -__builtin_inff());
      float alpha = dead ? 1.0f : __expf(mstate[i] - mnew);
      float rs = 0.0f;
#pragma unroll
      for (int nt = 0; nt < 4; ++nt) {
        float p = dead ? 0.0f : __expf(S[nt][i] - mnew);
        S[nt][i] = p;
        rs += p;
      }
      rs += __shfl_xor(rs, 1, 32);
      rs += __shfl_xor(rs, 2, 32);
      rs += __shfl_xor(rs, 4, 32);
      rs += __shfl_xor(rs, 8, 32);
      lstate[i] = lstate[i] * alpha + rs;
      mstate[i] = mnew;
#pragma unroll
      for (int nt2 = 0; nt2 < 4; ++nt2) O[nt2][i] *= alpha;
    }

    // restage P (C-layout -> LDS row-major); DS ops are in-order per wave.
#pragma unroll
    for (int nt = 0; nt < 4; ++nt)
#pragma unroll
      for (int i = 0; i < 8; ++i)
        Ps[w][i + 8 * half][nt * 16 + lane16] = (__bf16)S[nt][i];

    // O += P @ V : batch A + all 4 B fragments, then 4 back-to-back WMMAs
#pragma unroll
    for (int kk = 0; kk < 2; ++kk) {
      v16bf pa = frag_a_row(&Ps[w][lane16][kk * 32], half);
      v16bf vb[4];
#pragma unroll
      for (int nt2 = 0; nt2 < 4; ++nt2)
        vb[nt2] = frag_b_row(&Vst[nt2 * 16 + lane16][kk * 32], half);
#pragma unroll
      for (int nt2 = 0; nt2 < 4; ++nt2)
        O[nt2] = WMMA_BF16(pa, vb[nt2], O[nt2]);
    }
  }

  // normalize and store attention output as bf16 [B,N,C] (C = h*64 + d)
  __bf16* op = aw + (size_t)(b * Nseq + qbase + w * 16 + 8 * half) * Cdim + h * Dh;
#pragma unroll
  for (int nt2 = 0; nt2 < 4; ++nt2)
#pragma unroll
    for (int i = 0; i < 8; ++i) {
      float l = lstate[i];
      float v = (l > 0.0f) ? O[nt2][i] / l : 0.0f;
      op[(size_t)i * Cdim + nt2 * 16 + lane16] = (__bf16)v;
    }
}

// ---------------- kernel 4: output projection (bf16 WMMA GEMM, fp32 out) ----

__global__ __launch_bounds__(128)
void out_proj(const __bf16* __restrict__ aw, const float* __restrict__ Wo,
              const float* __restrict__ bo, float* __restrict__ out) {
  __shared__ __align__(16) __bf16 As [64][32];
  __shared__ __align__(16) __bf16 Bst[64][32];   // transposed

  int tid = threadIdx.x;
  int lane = tid & 31, w = tid >> 5;
  int lane16 = lane & 15, half = lane >> 4;
  int cblk = blockIdx.x * 64;
  int rblk = blockIdx.y * 64;

  int a_rb = tid >> 4, a_c2 = (tid & 15) * 2;
  int b_c  = tid & 63, b_r2 = tid >> 6;

  const __bf16* apg = &aw[(size_t)(rblk + a_rb) * Cdim + a_c2];
  const float*  bpg = &Wo[(size_t)(2 * b_r2) * Cdim + cblk + b_c];

  v8f acc[4] = {};

  uint32_t ta[8];
  float tb0[8], tb1[8];
#pragma unroll
  for (int it = 0; it < 8; ++it)
    ta[it] = *(const uint32_t*)&apg[(size_t)it * 8 * Cdim];
#pragma unroll
  for (int it = 0; it < 8; ++it) {
    tb0[it] = bpg[(size_t)(4 * it) * Cdim];
    tb1[it] = bpg[(size_t)(4 * it + 1) * Cdim];
  }

  for (int k0 = 0; k0 < Cdim; k0 += 32) {
    __syncthreads();
#pragma unroll
    for (int it = 0; it < 8; ++it)
      *(uint32_t*)&As[a_rb + it * 8][a_c2] = ta[it];
#pragma unroll
    for (int it = 0; it < 8; ++it)
      *(uint32_t*)&Bst[b_c][2 * b_r2 + 4 * it] = pack2bf(tb0[it], tb1[it]);
    __syncthreads();

    if (k0 + 32 < Cdim) {
      int kn = k0 + 32;
#pragma unroll
      for (int it = 0; it < 8; ++it)
        ta[it] = *(const uint32_t*)&apg[kn + (size_t)it * 8 * Cdim];
#pragma unroll
      for (int it = 0; it < 8; ++it) {
        tb0[it] = bpg[(size_t)(kn + 4 * it) * Cdim];
        tb1[it] = bpg[(size_t)(kn + 4 * it + 1) * Cdim];
      }
    }

    v16bf af = frag_a_row(&As[w * 16 + lane16][0], half);
    v16bf bf[4];
#pragma unroll
    for (int nt = 0; nt < 4; ++nt)
      bf[nt] = frag_b_row(&Bst[nt * 16 + lane16][0], half);
#pragma unroll
    for (int nt = 0; nt < 4; ++nt)
      acc[nt] = WMMA_BF16(af, bf[nt], acc[nt]);
  }

  float* op = out + (size_t)(rblk + w * 16 + 8 * half) * Cdim + cblk;
#pragma unroll
  for (int nt = 0; nt < 4; ++nt) {
    int c = nt * 16 + lane16;
    float bias = bo[cblk + c];
#pragma unroll
    for (int i = 0; i < 8; ++i)
      op[(size_t)i * Cdim + c] = acc[nt][i] + bias;
  }
}

// ---------------- launcher --------------------------------------------------

extern "C" void kernel_launch(void* const* d_in, const int* in_sizes, int n_in,
                              void* d_out, int out_size, void* d_ws, size_t ws_size,
                              hipStream_t stream) {
  const float* x    = (const float*)d_in[0];
  const int*   am   = (const int*)d_in[1];
  const float* Wqkv = (const float*)d_in[2];
  const float* bqkv = (const float*)d_in[3];
  const float* We   = (const float*)d_in[4];
  const float* be   = (const float*)d_in[5];
  const float* Wo   = (const float*)d_in[6];
  const float* bo   = (const float*)d_in[7];
  float* out = (float*)d_out;

  // workspace layout
  char* ws = (char*)d_ws;
  const size_t qkv_bytes = (size_t)3 * QKV_S_STRIDE * sizeof(__bf16);  // 25,165,824
  const size_t ew_bytes  = (size_t)Bsz * Nseq * Hn * sizeof(float);    //    262,144
  __bf16* qkvws = (__bf16*)ws;
  float*  ewws  = (float*)(ws + qkv_bytes);
  __bf16* awws  = (__bf16*)(ws + qkv_bytes + ew_bytes);                //  8,388,608

  gate_kernel<<<dim3((Bsz * Nseq * Hn) / 256), 256, 0, stream>>>(x, We, be, ewws);
  qkv_gemm<<<dim3(QKVN / 64, (Bsz * Nseq) / 64), 128, 0, stream>>>(x, Wqkv, bqkv, qkvws);
  attn_kernel<<<dim3(Bsz * Hn * (Nseq / 64)), 128, 0, stream>>>(qkvws, ewws, am, awws);
  out_proj<<<dim3(Cdim / 64, (Bsz * Nseq) / 64), 128, 0, stream>>>(awws, Wo, bo, out);
}
